// EGNNNBodyNetV2_13950053777583
// MI455X (gfx1250) — compile-verified, matrix operand order checked by reading, not run on hardware
//
#include <hip/hip_runtime.h>
#include <hip/hip_bf16.h>

typedef __attribute__((ext_vector_type(16))) _Float16 v16h;
typedef __attribute__((ext_vector_type(8)))  _Float16 v8h;
typedef __attribute__((ext_vector_type(8)))  float    v8f;

constexpr int kHID    = 128;
constexpr int kNP     = 5;
constexpr int kNL     = 4;
constexpr int kB      = 4096;
constexpr int kBT     = 16;               // batch elems per workgroup
constexpr int kNROWS  = kBT * kNP;        // 80 node rows
constexpr int kEROWS  = kBT * kNP * kNP;  // 400 edge rows
constexpr int kNT_N   = kNROWS / 16;      // 5 node row-tiles
constexpr int kNT_E   = kEROWS / 16;      // 25 edge row-tiles
constexpr int kTHREADS = 256;
constexpr int kNWAVES  = 8;

// ---- dynamic LDS layout (bytes) ----
constexpr int OFF_H    = 0;                       // hF32   [80][128] f32  40960
constexpr int OFF_P    = 40960;                   // Pbuf   [80][128] f32  40960
constexpr int OFF_Q    = 81920;                   // Qbuf   [80][128] f32  40960
constexpr int OFF_HN   = 122880;                  // hnH    [80][128] f16  20480
constexpr int OFF_AGG  = 143360;                  // aggH   [80][128] f16  20480
constexpr int OFF_EDGE = 163840;                  // edgeH  [400][128] f16 102400
constexpr int OFF_X    = 266240;                  // xF     [80][2]  f32   640
constexpr int OFF_D2   = 266880;                  // d2s    [400]    f32   1600
constexpr int OFF_CW   = 268480;                  // cws    [400]    f32   1600
constexpr int SMEM_BYTES = 270080;

struct LayerP {
  const float *att_b, *att_w, *e1_b, *e1_w, *e2_b, *f1_b, *f2_b,
              *h1_b, *h2_b, *norm_b, *norm_g, *x1_b, *x2_b, *x2_w;
  const _Float16 *W1a, *W1b, *We2, *Wx1, *Wh1, *Wh2, *Wf1, *Wf2;
};
struct NetP {
  const float *state, *embed, *enc_b, *enc_ln_b, *enc_ln_g;
  const float *dec1_b, *dec2_b, *dec2_w;
  const _Float16 *Wenc, *Wdec1;
  LayerP L[kNL];
  float* out;
};

union U8h  { v8h  v; _Float16 e[8]; };
union U16h { v16h v; _Float16 e[16]; };
union U8f  { v8f  v; float    e[8]; };

__device__ __forceinline__ float siluf(float x) { return x / (1.f + __expf(-x)); }

// Load a 16x32 f16 fragment (A layout; B uses the mirrored layout so the same
// loader works when the weight matrix is stored transposed [dout][din]).
// lane L holds row (L&15); K(t) = (t<8?0:16) + (L>>4)*8 + (t&7)  -> two
// contiguous 16-byte chunks per lane.
__device__ __forceinline__ v16h load_frag(const _Float16* base, int ld, int lane) {
  const int r  = lane & 15;
  const int kh = (lane >> 4) << 3;
  const _Float16* p = base + r * ld + kh;
  U8h lo, hi;
  lo.v = *(const v8h*)(p);
  hi.v = *(const v8h*)(p + 16);
  U16h out;
#pragma unroll
  for (int t = 0; t < 8; ++t) { out.e[t] = lo.e[t]; out.e[t + 8] = hi.e[t]; }
  return out.v;
}

// One 16-row tile x 128 output cols: acc[8] over 8 N-tiles, K contraction in
// steps of 32 via v_wmma_f32_16x16x32_f16. Wt is [dout][K] row-major f16.
__device__ __forceinline__ void gemm8(const _Float16* A, const _Float16* Wt, int K,
                                      v8f acc[8], int lane) {
  for (int ks = 0; ks < K; ks += 32) {
    v16h a = load_frag(A + ks, K, lane);
#pragma unroll
    for (int nt = 0; nt < 8; ++nt) {
      v16h b = load_frag(Wt + nt * 16 * K + ks, K, lane);
      acc[nt] = __builtin_amdgcn_wmma_f32_16x16x32_f16(
          false, a, false, b, (short)0, acc[nt], false, false);
    }
  }
}

__device__ __forceinline__ void init_bias(v8f acc[8], const float* b, int n0, int lane) {
  const int n = lane & 15;
#pragma unroll
  for (int nt = 0; nt < 8; ++nt) {
    float bv = b ? b[n0 + nt * 16 + n] : 0.f;
    U8f u;
#pragma unroll
    for (int r = 0; r < 8; ++r) u.e[r] = bv;
    acc[nt] = u.v;
  }
}

// C/D layout: acc[nt] lane holds row M = (lane>>4)*8 + r, col N = (lane&15)+nt*16
__device__ __forceinline__ void store8_f16(_Float16* out, int ldo, const v8f acc[8],
                                           int lane, bool dosilu) {
  const int n  = lane & 15;
  const int m0 = (lane >> 4) << 3;
#pragma unroll
  for (int nt = 0; nt < 8; ++nt) {
    U8f u; u.v = acc[nt];
#pragma unroll
    for (int r = 0; r < 8; ++r) {
      float v = u.e[r];
      if (dosilu) v = siluf(v);
      out[(m0 + r) * ldo + nt * 16 + n] = (_Float16)v;
    }
  }
}

__device__ __forceinline__ void store8_f32(float* out, int ldo, const v8f acc[8],
                                           int lane, bool dosilu) {
  const int n  = lane & 15;
  const int m0 = (lane >> 4) << 3;
#pragma unroll
  for (int nt = 0; nt < 8; ++nt) {
    U8f u; u.v = acc[nt];
#pragma unroll
    for (int r = 0; r < 8; ++r) {
      float v = u.e[r];
      if (dosilu) v = siluf(v);
      out[(m0 + r) * ldo + nt * 16 + n] = v;
    }
  }
}

__device__ __forceinline__ void ln_row_f32(const float* x, float* y,
                                           const float* g, const float* b) {
  float m = 0.f;
  for (int c = 0; c < kHID; ++c) m += x[c];
  m *= (1.f / kHID);
  float v = 0.f;
  for (int c = 0; c < kHID; ++c) { float d = x[c] - m; v += d * d; }
  v *= (1.f / kHID);
  const float inv = rsqrtf(v + 1e-5f);
  for (int c = 0; c < kHID; ++c) y[c] = (x[c] - m) * inv * g[c] + b[c];
}

__device__ __forceinline__ void ln_row_f16(const float* x, _Float16* y,
                                           const float* g, const float* b) {
  float m = 0.f;
  for (int c = 0; c < kHID; ++c) m += x[c];
  m *= (1.f / kHID);
  float v = 0.f;
  for (int c = 0; c < kHID; ++c) { float d = x[c] - m; v += d * d; }
  v *= (1.f / kHID);
  const float inv = rsqrtf(v + 1e-5f);
  for (int c = 0; c < kHID; ++c) y[c] = (_Float16)((x[c] - m) * inv * g[c] + b[c]);
}

// Weight prep: dst[n*dinPad + k] = (k<din) ? f16(src[k*dout + n]) : 0
__global__ void w_to_f16t(const float* __restrict__ src, _Float16* __restrict__ dst,
                          int din, int dout, int dinPad) {
  int idx = blockIdx.x * blockDim.x + threadIdx.x;
  if (idx >= dout * dinPad) return;
  int n = idx / dinPad, k = idx - n * dinPad;
  dst[idx] = (k < din) ? (_Float16)src[k * dout + n] : (_Float16)0.f;
}

__global__ __launch_bounds__(kTHREADS) void egnn_main(NetP NP) {
  extern __shared__ char smem[];
  float*    hF32  = (float*)(smem + OFF_H);
  float*    Pbuf  = (float*)(smem + OFF_P);
  float*    Qbuf  = (float*)(smem + OFF_Q);
  _Float16* hnH   = (_Float16*)(smem + OFF_HN);
  _Float16* aggH  = (_Float16*)(smem + OFF_AGG);
  _Float16* edgeH = (_Float16*)(smem + OFF_EDGE);
  float*    xF    = (float*)(smem + OFF_X);
  float*    d2s   = (float*)(smem + OFF_D2);
  float*    cws   = (float*)(smem + OFF_CW);

  const int tid   = threadIdx.x;
  const int lane  = tid & 31;
  const int wave  = tid >> 5;
  const int bBase = blockIdx.x * kBT;

  // ---------------- encoder input: [state(4) | emb(64) | 0-pad] ----------------
  for (int idx = tid; idx < kNROWS * kHID; idx += kTHREADS) {
    int n = idx >> 7, c = idx & 127;
    int b = bBase + n / kNP, i = n % kNP;
    float v = 0.f;
    if (c < 4)       v = NP.state[(b * kNP + i) * 4 + c];
    else if (c < 68) v = NP.embed[i * 64 + (c - 4)];
    edgeH[idx] = (_Float16)v;
  }
  for (int idx = tid; idx < kNROWS * 2; idx += kTHREADS) {
    int n = idx >> 1, c = idx & 1;
    int b = bBase + n / kNP, i = n % kNP;
    xF[idx] = NP.state[(b * kNP + i) * 4 + c];
  }
  __syncthreads();

  // encoder GEMM -> silu -> Pbuf (pre-LN, f32)
  for (int t = wave; t < kNT_N; t += kNWAVES) {
    v8f acc[8]; init_bias(acc, NP.enc_b, 0, lane);
    gemm8(edgeH + t * 16 * kHID, NP.Wenc, kHID, acc, lane);
    store8_f32(Pbuf + t * 16 * kHID, kHID, acc, lane, true);
  }
  __syncthreads();
  if (tid < kNROWS)
    ln_row_f32(Pbuf + tid * kHID, hF32 + tid * kHID, NP.enc_ln_g, NP.enc_ln_b);
  __syncthreads();

  // ---------------- layers ----------------
  for (int l = 0; l < kNL; ++l) {
    const LayerP& L = NP.L[l];

    // hn = LN(h)
    if (tid < kNROWS)
      ln_row_f16(hF32 + tid * kHID, hnH + tid * kHID, L.norm_g, L.norm_b);
    __syncthreads();

    // P = hn@W1a + b_e1 ; Q = hn@W1b   (10 tile jobs)
    for (int job = wave; job < 2 * kNT_N; job += kNWAVES) {
      int t = job % kNT_N;
      bool isP = job < kNT_N;
      v8f acc[8]; init_bias(acc, isP ? L.e1_b : nullptr, 0, lane);
      gemm8(hnH + t * 16 * kHID, isP ? L.W1a : L.W1b, kHID, acc, lane);
      store8_f32((isP ? Pbuf : Qbuf) + t * 16 * kHID, kHID, acc, lane, false);
    }
    __syncthreads();

    // d2 per edge
    for (int r = tid; r < kEROWS; r += kTHREADS) {
      int bl = r / 25, rem = r % 25, i = rem / 5, j = rem % 5;
      float dx = xF[(bl * kNP + i) * 2 + 0] - xF[(bl * kNP + j) * 2 + 0];
      float dy = xF[(bl * kNP + i) * 2 + 1] - xF[(bl * kNP + j) * 2 + 1];
      d2s[r] = dx * dx + dy * dy;
    }
    __syncthreads();

    // u = silu(P_i + Q_j + d2*w1c)  -> edgeH
    const float* w1c = L.e1_w + 256 * kHID;  // last row of e1.w
    for (int idx = tid; idx < kEROWS * kHID; idx += kTHREADS) {
      int r = idx >> 7, c = idx & 127;
      int bl = r / 25, rem = r % 25, i = rem / 5, j = rem % 5;
      float v = Pbuf[(bl * kNP + i) * kHID + c] + Qbuf[(bl * kNP + j) * kHID + c]
              + d2s[r] * w1c[c];
      edgeH[idx] = (_Float16)siluf(v);
    }
    __syncthreads();

    // m = silu(u @ We2 + b)  (in-place)
    for (int t = wave; t < kNT_E; t += kNWAVES) {
      v8f acc[8]; init_bias(acc, L.e2_b, 0, lane);
      gemm8(edgeH + t * 16 * kHID, L.We2, kHID, acc, lane);
      store8_f16(edgeH + t * 16 * kHID, kHID, acc, lane, true);
    }
    __syncthreads();

    // attention gate: m *= sigmoid(m . w_att + b)
    for (int r = tid; r < kEROWS; r += kTHREADS) {
      _Float16* row = edgeH + r * kHID;
      float s = L.att_b[0];
      for (int c = 0; c < kHID; ++c) s += (float)row[c] * L.att_w[c];
      float g = 1.f / (1.f + __expf(-s));
      for (int c = 0; c < kHID; ++c) row[c] = (_Float16)((float)row[c] * g);
    }
    __syncthreads();

    // agg[n] = sum_{j != i} m
    for (int idx = tid; idx < kNROWS * kHID; idx += kTHREADS) {
      int n = idx >> 7, c = idx & 127;
      int bl = n / kNP, i = n % kNP;
      float s = 0.f;
      for (int j = 0; j < kNP; ++j)
        if (j != i) s += (float)edgeH[(bl * 25 + i * 5 + j) * kHID + c];
      aggH[idx] = (_Float16)s;
    }
    __syncthreads();

    // v = silu(m @ Wx1 + b)  (in-place)
    for (int t = wave; t < kNT_E; t += kNWAVES) {
      v8f acc[8]; init_bias(acc, L.x1_b, 0, lane);
      gemm8(edgeH + t * 16 * kHID, L.Wx1, kHID, acc, lane);
      store8_f16(edgeH + t * 16 * kHID, kHID, acc, lane, true);
    }
    __syncthreads();

    // cw = v . w_x2 + b
    for (int r = tid; r < kEROWS; r += kTHREADS) {
      const _Float16* row = edgeH + r * kHID;
      float s = L.x2_b[0];
      for (int c = 0; c < kHID; ++c) s += (float)row[c] * L.x2_w[c];
      cws[r] = s;
    }
    __syncthreads();

    // x += sum_{j!=i} diff*cw / (N-1)
    float nx = 0.f, ny = 0.f;
    if (tid < kNROWS) {
      int bl = tid / kNP, i = tid % kNP;
      float fx = xF[tid * 2], fy = xF[tid * 2 + 1], sx = 0.f, sy = 0.f;
      for (int j = 0; j < kNP; ++j)
        if (j != i) {
          float c = cws[bl * 25 + i * 5 + j];
          sx += (fx - xF[(bl * kNP + j) * 2 + 0]) * c;
          sy += (fy - xF[(bl * kNP + j) * 2 + 1]) * c;
        }
      nx = fx + sx * 0.25f;
      ny = fy + sy * 0.25f;
    }
    __syncthreads();
    if (tid < kNROWS) { xF[tid * 2] = nx; xF[tid * 2 + 1] = ny; }
    __syncthreads();

    // cat = [hn | agg] (f16) in edgeH; t1 behind it
    _Float16* catH = edgeH;                 // [80][256]
    _Float16* t1H  = edgeH + kNROWS * 256;  // [80][128]
    for (int idx = tid; idx < kNROWS * 256; idx += kTHREADS) {
      int n = idx >> 8, c = idx & 255;
      catH[idx] = (c < kHID) ? hnH[n * kHID + c] : aggH[n * kHID + (c - kHID)];
    }
    __syncthreads();

    // t1 = silu(cat @ Wh1 + b)   K=256
    for (int t = wave; t < kNT_N; t += kNWAVES) {
      v8f acc[8]; init_bias(acc, L.h1_b, 0, lane);
      gemm8(catH + t * 16 * 256, L.Wh1, 256, acc, lane);
      store8_f16(t1H + t * 16 * kHID, kHID, acc, lane, true);
    }
    __syncthreads();

    // h += hn + (t1 @ Wh2 + b)
    for (int t = wave; t < kNT_N; t += kNWAVES) {
      v8f acc[8]; init_bias(acc, L.h2_b, 0, lane);
      gemm8(t1H + t * 16 * kHID, L.Wh2, kHID, acc, lane);
      const int n = lane & 15, m0 = (lane >> 4) << 3;
#pragma unroll
      for (int nt = 0; nt < 8; ++nt) {
        U8f u; u.v = acc[nt];
#pragma unroll
        for (int r = 0; r < 8; ++r) {
          int row = t * 16 + m0 + r, col = nt * 16 + n;
          hF32[row * kHID + col] += (float)hnH[row * kHID + col] + u.e[r];
        }
      }
    }
    __syncthreads();

    // feed-forward block: h += f2(silu(f1(h)))
    for (int idx = tid; idx < kNROWS * kHID; idx += kTHREADS)
      hnH[idx] = (_Float16)hF32[idx];
    __syncthreads();
    _Float16* fmid = edgeH;  // [80][256]
    for (int job = wave; job < 2 * kNT_N; job += kNWAVES) {
      int t = job % kNT_N, half = job / kNT_N;
      v8f acc[8]; init_bias(acc, L.f1_b, half * 128, lane);
      gemm8(hnH + t * 16 * kHID, L.Wf1 + (half * 128) * kHID, kHID, acc, lane);
      store8_f16(fmid + t * 16 * 256 + half * 128, 256, acc, lane, true);
    }
    __syncthreads();
    for (int t = wave; t < kNT_N; t += kNWAVES) {
      v8f acc[8]; init_bias(acc, L.f2_b, 0, lane);
      gemm8(fmid + t * 16 * 256, L.Wf2, 256, acc, lane);
      const int n = lane & 15, m0 = (lane >> 4) << 3;
#pragma unroll
      for (int nt = 0; nt < 8; ++nt) {
        U8f u; u.v = acc[nt];
#pragma unroll
        for (int r = 0; r < 8; ++r) {
          int row = t * 16 + m0 + r, col = nt * 16 + n;
          hF32[row * kHID + col] += u.e[r];
        }
      }
    }
    __syncthreads();
  }  // layers

  // ---------------- decoder ----------------
  for (int idx = tid; idx < kNROWS * kHID; idx += kTHREADS)
    hnH[idx] = (_Float16)hF32[idx];
  __syncthreads();
  for (int t = wave; t < kNT_N; t += kNWAVES) {
    v8f acc[8]; init_bias(acc, NP.dec1_b, 0, lane);
    gemm8(hnH + t * 16 * kHID, NP.Wdec1, kHID, acc, lane);
    store8_f16(edgeH + t * 16 * kHID, kHID, acc, lane, true);
  }
  __syncthreads();
  // dec2: dout = 4 -> per-row dots; out = state + delta
  for (int job = tid; job < kNROWS * 4; job += kTHREADS) {
    int n = job >> 2, c = job & 3;
    int b = bBase + n / kNP, i = n % kNP;
    float s = NP.dec2_b[c];
    const _Float16* row = edgeH + n * kHID;
    for (int k = 0; k < kHID; ++k) s += (float)row[k] * NP.dec2_w[k * 4 + c];
    int gi = (b * kNP + i) * 4 + c;
    NP.out[gi] = NP.state[gi] + s;
  }
}

extern "C" void kernel_launch(void* const* d_in, const int* in_sizes, int n_in,
                              void* d_out, int out_size, void* d_ws, size_t ws_size,
                              hipStream_t stream) {
  (void)in_sizes; (void)n_in; (void)out_size; (void)ws_size;

  // JAX tree-flatten order (dict keys sorted):
  // dec1{b,w}, dec2{b,w}, embed, enc_lin{b,w}, enc_ln{b,g},
  // layers[l]{att{b,w}, e1{b,w}, e2{b,w}, f1{b,w}, f2{b,w},
  //           h1{b,w}, h2{b,w}, norm{b,g}, x1{b,w}, x2{b,w}}, state
  NetP NPv;
  NPv.dec1_b   = (const float*)d_in[0];
  const float* dec1_w = (const float*)d_in[1];
  NPv.dec2_b   = (const float*)d_in[2];
  NPv.dec2_w   = (const float*)d_in[3];
  NPv.embed    = (const float*)d_in[4];
  NPv.enc_b    = (const float*)d_in[5];
  const float* enc_w = (const float*)d_in[6];
  NPv.enc_ln_b = (const float*)d_in[7];
  NPv.enc_ln_g = (const float*)d_in[8];
  NPv.state    = (const float*)d_in[89];
  NPv.out      = (float*)d_out;

  _Float16* ws = (_Float16*)d_ws;
  size_t off = 0;
  auto alloc = [&](size_t nh) { _Float16* p = ws + off; off += nh; return p; };
  auto T = [&](const float* src, _Float16* dst, int din, int dout, int dinPad) {
    int total = dout * dinPad;
    w_to_f16t<<<(total + 255) / 256, 256, 0, stream>>>(src, dst, din, dout, dinPad);
  };

  _Float16* Wenc = alloc(128 * 128);
  _Float16* Wdec1 = alloc(128 * 128);
  T(enc_w, Wenc, 68, 128, 128);   // pad din 68 -> 128 with zeros
  T(dec1_w, Wdec1, 128, 128, 128);
  NPv.Wenc = Wenc; NPv.Wdec1 = Wdec1;

  for (int l = 0; l < kNL; ++l) {
    int base = 9 + 20 * l;
    LayerP& L = NPv.L[l];
    L.att_b  = (const float*)d_in[base + 0];
    L.att_w  = (const float*)d_in[base + 1];
    L.e1_b   = (const float*)d_in[base + 2];
    L.e1_w   = (const float*)d_in[base + 3];
    L.e2_b   = (const float*)d_in[base + 4];
    const float* e2_w = (const float*)d_in[base + 5];
    L.f1_b   = (const float*)d_in[base + 6];
    const float* f1_w = (const float*)d_in[base + 7];
    L.f2_b   = (const float*)d_in[base + 8];
    const float* f2_w = (const float*)d_in[base + 9];
    L.h1_b   = (const float*)d_in[base + 10];
    const float* h1_w = (const float*)d_in[base + 11];
    L.h2_b   = (const float*)d_in[base + 12];
    const float* h2_w = (const float*)d_in[base + 13];
    L.norm_b = (const float*)d_in[base + 14];
    L.norm_g = (const float*)d_in[base + 15];
    L.x1_b   = (const float*)d_in[base + 16];
    const float* x1_w = (const float*)d_in[base + 17];
    L.x2_b   = (const float*)d_in[base + 18];
    L.x2_w   = (const float*)d_in[base + 19];

    _Float16* W1a = alloc(128 * 128); T(L.e1_w,             W1a, 128, 128, 128);
    _Float16* W1b = alloc(128 * 128); T(L.e1_w + 128 * 128, W1b, 128, 128, 128);
    _Float16* We2 = alloc(128 * 128); T(e2_w, We2, 128, 128, 128);
    _Float16* Wx1 = alloc(128 * 128); T(x1_w, Wx1, 128, 128, 128);
    _Float16* Wh1 = alloc(128 * 256); T(h1_w, Wh1, 256, 128, 256);
    _Float16* Wh2 = alloc(128 * 128); T(h2_w, Wh2, 128, 128, 128);
    _Float16* Wf1 = alloc(256 * 128); T(f1_w, Wf1, 128, 256, 128);
    _Float16* Wf2 = alloc(128 * 256); T(f2_w, Wf2, 256, 128, 256);
    L.W1a = W1a; L.W1b = W1b; L.We2 = We2; L.Wx1 = Wx1;
    L.Wh1 = Wh1; L.Wh2 = Wh2; L.Wf1 = Wf1; L.Wf2 = Wf2;
  }

  hipFuncSetAttribute(reinterpret_cast<const void*>(egnn_main),
                      hipFuncAttributeMaxDynamicSharedMemorySize, SMEM_BYTES);
  egnn_main<<<dim3(kB / kBT), dim3(kTHREADS), SMEM_BYTES, stream>>>(NPv);
}
